// SiameseGNN_2551210573858
// MI455X (gfx1250) — matrix operand order repaired and checked by reading, not compiled, not run on hardware
//
#include <hip/hip_runtime.h>
#include <hip/hip_bf16.h>
#include <math.h>

#define Nn 100000
#define Ee 1600000
#define Gg 64
#define Cc 128
#define Oo 64

typedef __attribute__((ext_vector_type(16))) _Float16 v16h;
typedef __attribute__((ext_vector_type(8)))  _Float16 v8h;
typedef __attribute__((ext_vector_type(8)))  float    v8f;

// ---------------- degree / norm precompute ----------------

__global__ void k_deg_init(float* __restrict__ deg) {
    int i = blockIdx.x * blockDim.x + threadIdx.x;
    if (i < Nn) deg[i] = 1.0f;   // self-loop weight 1
}

__global__ void k_deg_edge(const int* __restrict__ di, const float* __restrict__ w,
                           float* __restrict__ deg) {
    int e = blockIdx.x * blockDim.x + threadIdx.x;
    if (e < Ee) atomicAdd(&deg[di[e]], w[e]);
}

__global__ void k_deg_rsqrt(float* __restrict__ deg) {
    int i = blockIdx.x * blockDim.x + threadIdx.x;
    if (i < Nn) {
        float d = deg[i];
        deg[i] = (d > 0.0f) ? rsqrtf(d) : 0.0f;   // deg buffer now holds dinv
    }
}

__global__ void k_norm(const int* __restrict__ si, const int* __restrict__ di,
                       const float* __restrict__ w, const float* __restrict__ dinv,
                       float* __restrict__ norm) {
    int e = blockIdx.x * blockDim.x + threadIdx.x;
    if (e < Ee) norm[e] = dinv[si[e]] * w[e] * dinv[di[e]];
}

// ---------------- one-time weight pack: f32 row-major -> f16 B-fragment layout ----
// Bp[((L*8 + colTile)*4 + kk)*32 + lane][16 halves], where for W element (k,n):
//   colTile = n/16, r = n%16, kk = k/32, kin = k%32, lane = ((kin/8)%2)*16 + r,
//   element = kin%8 + (kin/16)*8   (matches ISA 16-bit 32x16 B layout: lanes16-31 hold K+8).
__global__ void k_packW(const float* __restrict__ W6, _Float16* __restrict__ Bp) {
    int idx = blockIdx.x * blockDim.x + threadIdx.x;   // 6*128*128
    if (idx >= 6 * Cc * Cc) return;
    int L   = idx >> 14;
    int rem = idx & 16383;
    int k   = rem >> 7;
    int n   = rem & 127;
    int colTile = n >> 4, r = n & 15;
    int kk = k >> 5, kin = k & 31;
    int lane = (((kin >> 3) & 1) << 4) + r;
    int elem = (kin & 7) + ((kin >> 4) << 3);
    size_t off = ((((size_t)L * 8 + colTile) * 4 + kk) * 32 + lane) * 16 + elem;
    Bp[off] = (_Float16)W6[idx];
}

// ---------------- layer-0 input: f32 -> f16 copy ----------------
__global__ void k_f32_to_f16(const float* __restrict__ x, _Float16* __restrict__ xh) {
    long idx = (long)blockIdx.x * blockDim.x + threadIdx.x;
    if (idx < (long)Nn * Cc) xh[idx] = (_Float16)x[idx];
}

// ---------------- dense x @ W via WMMA (f16 operands, f32 acc) ----------------
// Grid: Nn/16 blocks of 256 threads (8 waves). Wave w -> 16x16 tile at
// (rowTile = blockIdx.x, colTile = w). K = 128 -> 4 WMMA steps.
// A fragment: two contiguous 16B half-groups per lane (ISA A layout, hi lanes +8).
// B fragment: one contiguous 32B v16h from the pre-packed Bp.
__global__ void k_gemm_wmma(const _Float16* __restrict__ Xh,
                            const _Float16* __restrict__ BpL,
                            float* __restrict__ Y) {
    const int wave = threadIdx.x >> 5;
    const int lane = threadIdx.x & 31;
    const int row0 = blockIdx.x * 16;
    const int r    = lane & 15;
    const int hi   = lane >> 4;

    const _Float16* __restrict__ xrow  = Xh + (size_t)(row0 + r) * Cc + hi * 8;
    const _Float16* __restrict__ bbase = BpL + ((size_t)wave * 4) * 512 + lane * 16;

    v8f acc = {};
#pragma unroll
    for (int kk = 0; kk < 4; ++kk) {
        v8h alo = *(const v8h*)(xrow + kk * 32);        // K = kk*32 + hi*8 .. +7
        v8h ahi = *(const v8h*)(xrow + kk * 32 + 16);   // K = kk*32 + 16 + hi*8 .. +7
        v16h a = __builtin_shufflevector(alo, ahi,
                                         0, 1, 2, 3, 4, 5, 6, 7,
                                         8, 9, 10, 11, 12, 13, 14, 15);
        v16h b = *(const v16h*)(bbase + kk * 512);
        acc = __builtin_amdgcn_wmma_f32_16x16x32_f16(
            /*neg_a=*/false, a, /*neg_b=*/false, b,
            /*c_mod=*/(short)0, acc, /*reuse_a=*/false, /*reuse_b=*/false);
    }
#pragma unroll
    for (int v = 0; v < 8; ++v) {
        int row = row0 + hi * 8 + v;       // C/D layout: VGPR v -> M=v / M=8+v
        Y[(size_t)row * Cc + wave * 16 + r] = acc[v];
    }
}

// ---------------- self-loop init: xout = dinv^2 * xw ----------------
__global__ void k_self_init(const float* __restrict__ dinv, const float* __restrict__ xw,
                            float* __restrict__ xout) {
    long idx = (long)blockIdx.x * blockDim.x + threadIdx.x;
    if (idx >= (long)Nn * 32) return;
    int node = (int)(idx >> 5);
    int c4   = (int)(idx & 31);
    float dv = dinv[node];
    float sn = dv * dv;
    float4 v = ((const float4*)(xw + (size_t)node * Cc))[c4];
    float4 o;
    o.x = sn * v.x; o.y = sn * v.y; o.z = sn * v.z; o.w = sn * v.w;
    ((float4*)(xout + (size_t)node * Cc))[c4] = o;
}

// ---------------- edge scatter: xout[d] += norm[e] * xw[s] ----------------
// wave32-per-edge: 32 lanes x float4 = all 128 channels of one row.
__global__ void k_edge_scatter(const int* __restrict__ si, const int* __restrict__ di,
                               const float* __restrict__ norm, const float* __restrict__ xw,
                               float* __restrict__ xout) {
    const int lane = threadIdx.x & 31;
    const int wpb  = blockDim.x >> 5;
    const int gw   = blockIdx.x * wpb + (threadIdx.x >> 5);
    const int strd = gridDim.x * wpb;
    for (int e = gw; e < Ee; e += strd) {
        int nxt = e + strd;
        if (nxt < Ee && lane == 0)
            __builtin_prefetch(xw + (size_t)si[nxt] * Cc, 0, 1);  // global_prefetch_b8
        int s = si[e], d = di[e];
        float nv = norm[e];
        float4 v = ((const float4*)(xw + (size_t)s * Cc))[lane];
        float* dp = xout + (size_t)d * Cc + lane * 4;
        atomicAdd(dp + 0, nv * v.x);
        atomicAdd(dp + 1, nv * v.y);
        atomicAdd(dp + 2, nv * v.z);
        atomicAdd(dp + 3, nv * v.w);
    }
}

// ---------------- bias + optional tanh; also emits the f16 copy ----------------
__global__ void k_bias_act(float* __restrict__ x, _Float16* __restrict__ xh,
                           const float* __restrict__ bias, int act) {
    long idx = (long)blockIdx.x * blockDim.x + threadIdx.x;
    if (idx >= (long)Nn * Cc) return;
    int c = (int)(idx & (Cc - 1));
    float v = x[idx] + bias[c];
    v = act ? tanhf(v) : v;
    x[idx]  = v;
    xh[idx] = (_Float16)v;
}

// ---------------- global mean pool ----------------
__global__ void k_pool_zero(float* __restrict__ pool, float* __restrict__ cnt) {
    int i = blockIdx.x * blockDim.x + threadIdx.x;
    if (i < Gg * Cc) pool[i] = 0.0f;
    if (i < Gg) cnt[i] = 0.0f;
}

__global__ void k_pool_sum(const float* __restrict__ x, const int* __restrict__ batch,
                           float* __restrict__ pool, float* __restrict__ cnt) {
    long idx = (long)blockIdx.x * blockDim.x + threadIdx.x;
    if (idx >= (long)Nn * 32) return;
    int node = (int)(idx >> 5);
    int c4   = (int)(idx & 31);
    int g = batch[node];
    float4 v = ((const float4*)(x + (size_t)node * Cc))[c4];
    float* p = pool + (size_t)g * Cc + c4 * 4;
    atomicAdd(p + 0, v.x);
    atomicAdd(p + 1, v.y);
    atomicAdd(p + 2, v.z);
    atomicAdd(p + 3, v.w);
    if (c4 == 0) atomicAdd(cnt + g, 1.0f);
}

__global__ void k_pool_mean(const float* __restrict__ pool, const float* __restrict__ cnt,
                            float* __restrict__ h) {
    int i = blockIdx.x * blockDim.x + threadIdx.x;
    if (i >= Gg * Cc) return;
    int g = i >> 7;
    h[i] = pool[i] / fmaxf(cnt[g], 1.0f);
}

// ---------------- small FC + relu ----------------
__global__ void k_fc_relu(const float* __restrict__ h, const float* __restrict__ Wm,
                          const float* __restrict__ b, float* __restrict__ out,
                          int K, int Ncol) {
    int idx = blockIdx.x * blockDim.x + threadIdx.x;
    if (idx >= Gg * Ncol) return;
    int r = idx / Ncol, c = idx - r * Ncol;
    float acc = b[c];
    for (int k = 0; k < K; ++k) acc += h[r * K + k] * Wm[k * Ncol + c];
    out[idx] = fmaxf(acc, 0.0f);
}

// ---------------- final L2 distance per graph ----------------
__global__ void k_l2dist(const float* __restrict__ h1, const float* __restrict__ h2,
                         float* __restrict__ out) {
    int g = blockIdx.x * blockDim.x + threadIdx.x;
    if (g >= Gg) return;
    float s = 0.0f;
    for (int c = 0; c < Oo; ++c) {
        float d = h1[g * Oo + c] - h2[g * Oo + c];
        s += d * d;
    }
    out[g] = sqrtf(s);
}

// ======================================================================

extern "C" void kernel_launch(void* const* d_in, const int* in_sizes, int n_in,
                              void* d_out, int out_size, void* d_ws, size_t ws_size,
                              hipStream_t stream) {
    (void)in_sizes; (void)n_in; (void)out_size; (void)ws_size;

    const float* x1    = (const float*)d_in[0];
    const int*   ei1   = (const int*)d_in[1];
    const float* ea1   = (const float*)d_in[2];
    const int*   b1    = (const int*)d_in[3];
    const float* x2    = (const float*)d_in[4];
    const int*   ei2   = (const int*)d_in[5];
    const float* ea2   = (const float*)d_in[6];
    const int*   b2    = (const int*)d_in[7];
    const float* convW = (const float*)d_in[8];
    const float* convB = (const float*)d_in[9];
    const float* fcW1  = (const float*)d_in[10];
    const float* fcb1  = (const float*)d_in[11];
    const float* fcW   = (const float*)d_in[12];
    const float* fcb   = (const float*)d_in[13];

    float* wsf   = (float*)d_ws;
    float* deg   = wsf;                        // N (becomes dinv)
    float* norm  = deg + Nn;                   // E
    float* xw    = norm + Ee;                  // N*C
    float* xa    = xw + (size_t)Nn * Cc;       // N*C
    float* xb    = xa + (size_t)Nn * Cc;       // N*C
    float* pool  = xb + (size_t)Nn * Cc;       // G*C
    float* cnt   = pool + Gg * Cc;             // G
    float* hmean = cnt + Gg;                   // G*C
    float* hA    = hmean + Gg * Cc;            // G*O
    float* hB    = hA + Gg * Oo;               // G*O
    float* h1    = hB + Gg * Oo;               // G*O
    float* h2    = h1 + Gg * Oo;               // G*O
    _Float16* xh = (_Float16*)(h2 + Gg * Oo);  // N*C halves (f16 activations)
    _Float16* Bp = xh + (size_t)Nn * Cc;       // 6*128*128 halves (packed weights)

    const int acts[6] = {1, 1, 0, 1, 1, 1};

    // pack all 6 conv weights into WMMA B-fragment layout (shared by both towers)
    k_packW<<<(6 * Cc * Cc + 255) / 256, 256, 0, stream>>>(convW, Bp);

    auto tower = [&](const float* x, const int* ei, const float* ea,
                     const int* batch, float* hout) {
        const int* si = ei;
        const int* di = ei + Ee;

        k_deg_init <<<(Nn + 255) / 256, 256, 0, stream>>>(deg);
        k_deg_edge <<<(Ee + 255) / 256, 256, 0, stream>>>(di, ea, deg);
        k_deg_rsqrt<<<(Nn + 255) / 256, 256, 0, stream>>>(deg);
        k_norm     <<<(Ee + 255) / 256, 256, 0, stream>>>(si, di, ea, deg, norm);

        k_f32_to_f16<<<(int)(((long)Nn * Cc + 255) / 256), 256, 0, stream>>>(x, xh);

        const float* cur = x;
        for (int L = 0; L < 6; ++L) {
            float* nxt = (L & 1) ? xb : xa;
            k_gemm_wmma   <<<Nn / 16, 256, 0, stream>>>(xh, Bp + (size_t)L * Cc * Cc, xw);
            k_self_init   <<<(Nn * 32 + 255) / 256, 256, 0, stream>>>(deg, xw, nxt);
            k_edge_scatter<<<20000, 256, 0, stream>>>(si, di, norm, xw, nxt);
            k_bias_act    <<<(int)(((long)Nn * Cc + 255) / 256), 256, 0, stream>>>(
                nxt, xh, convB + (size_t)L * Cc, acts[L]);
            cur = nxt;
        }

        k_pool_zero<<<(Gg * Cc + 255) / 256, 256, 0, stream>>>(pool, cnt);
        k_pool_sum <<<(Nn * 32 + 255) / 256, 256, 0, stream>>>(cur, batch, pool, cnt);
        k_pool_mean<<<(Gg * Cc + 255) / 256, 256, 0, stream>>>(pool, cnt, hmean);

        k_fc_relu<<<(Gg * Oo + 255) / 256, 256, 0, stream>>>(hmean, fcW1, fcb1, hA, Cc, Oo);
        const float* fin = hA;
        for (int i = 0; i < 5; ++i) {
            float* dst = (i == 4) ? hout : ((i & 1) ? hA : hB);
            k_fc_relu<<<(Gg * Oo + 255) / 256, 256, 0, stream>>>(
                fin, fcW + (size_t)i * Oo * Oo, fcb + (size_t)i * Oo, dst, Oo, Oo);
            fin = dst;
        }
    };

    tower(x1, ei1, ea1, b1, h1);
    tower(x2, ei2, ea2, b2, h2);
    k_l2dist<<<1, 64, 0, stream>>>(h1, h2, (float*)d_out);
}
